// Attention_15676630630889
// MI455X (gfx1250) — matrix operand order repaired and checked by reading, not compiled, not run on hardware
//
#include <hip/hip_runtime.h>
#include <hip/hip_bf16.h>
#include <cstdint>
#include <cstddef>

// Problem constants (match reference)
constexpr int BB   = 4;
constexpr int SS   = 1024;
constexpr int DD   = 2048;
constexpr int NH   = 32;
constexpr int NKV  = 8;
constexpr int HD   = 64;
constexpr int HALF = HD / 2;

typedef __attribute__((ext_vector_type(16))) _Float16 v16h;
typedef __attribute__((ext_vector_type(8)))  _Float16 v8h;
typedef __attribute__((ext_vector_type(8)))  float    v8f;
typedef __attribute__((ext_vector_type(4)))  float    v4f;

__device__ __forceinline__ v16h make_frag(const v8h lo, const v8h hi) {
  v16h r;
#pragma unroll
  for (int i = 0; i < 8; ++i) { r[i] = lo[i]; r[i + 8] = hi[i]; }
  return r;
}

// CDNA5 async global->LDS copy (ASYNCcnt-tracked, bypasses VGPRs).
// One lane moves 16 bytes; LDS address comes from the low 32 bits of the
// generic pointer (LDS aperture: addr[31:0] = LDS offset).
__device__ __forceinline__ void async_ld_b128(void* lds_dst, const void* gsrc) {
  uint32_t l = (uint32_t)(uintptr_t)lds_dst;
  unsigned long long g = (unsigned long long)(uintptr_t)gsrc;
  asm volatile("global_load_async_to_lds_b128 %0, %1, off"
               :: "v"(l), "v"(g) : "memory");
}
__device__ __forceinline__ void wait_async0() {
  asm volatile("s_wait_asynccnt 0x0" ::: "memory");
}

// ---------------------------------------------------------------------------
// fp32 -> f16 cast (vectorized x4)
// ---------------------------------------------------------------------------
__global__ __launch_bounds__(256) void cast_f32_to_f16(const float* __restrict__ in,
                                                       _Float16* __restrict__ out,
                                                       int n4) {
  int i = blockIdx.x * blockDim.x + threadIdx.x;
  if (i < n4) {
    v4f f = *(const v4f*)(in + (size_t)i * 4);
    _Float16* o = out + (size_t)i * 4;
    o[0] = (_Float16)f[0]; o[1] = (_Float16)f[1];
    o[2] = (_Float16)f[2]; o[3] = (_Float16)f[3];
  }
}

// ---------------------------------------------------------------------------
// Tiled WMMA GEMM: C[M,N] = A[M,K](f16,row-major) * B[K,N](f16,row-major)
// Block tile 128x64, 256 threads = 8 waves; wave w owns rows w*16..w*16+15
// and all 64 columns (4 WMMAs per K-step). A tile staged via async
// global->LDS b128; B staged transposed through VGPRs so B-fragments
// (lane n = l&15, K = (l>>4)*8 + {0..7} U 16 + (l>>4)*8 + {0..7}) are two 8-half LDS reads.
// M % 128 == 0, N % 64 == 0, K % 32 == 0 (true for all uses here).
// ---------------------------------------------------------------------------
template <typename OutT>
__global__ __launch_bounds__(256) void wmma_gemm(const _Float16* __restrict__ A,
                                                 const _Float16* __restrict__ Bm,
                                                 OutT* __restrict__ C,
                                                 int M, int N, int K) {
  __shared__ __align__(16) _Float16 As[128][32];  // 8 KB, row-major
  __shared__ __align__(16) _Float16 Bs[64][40];   // 5 KB, [n][k], 80B rows (16B align)

  const int tid   = threadIdx.x;
  const int lane  = tid & 31;
  const int wid   = tid >> 5;          // 0..7 : 16-row strip
  const int m0    = blockIdx.y * 128;
  const int n0    = blockIdx.x * 64;
  const int mfrag = lane & 15;
  const int hh    = lane >> 4;

  const int arow = tid >> 1, acol = (tid & 1) * 16; // A staging: 16 halves/thread
  const int brow = tid / 8,  bcg  = (tid % 8) * 8;  // B staging: 8 halves/thread

  v8f acc[4] = {};

  for (int k0 = 0; k0 < K; k0 += 32) {
    __syncthreads();
    // stage A tile: two async b128 per thread, straight global->LDS
    const _Float16* ag = &A[(size_t)(m0 + arow) * K + k0 + acol];
    async_ld_b128(&As[arow][acol],     ag);
    async_ld_b128(&As[arow][acol + 8], ag + 8);
    // stage B tile transposed: Bs[n][k]
    v8h bv = *(const v8h*)&Bm[(size_t)(k0 + brow) * N + n0 + bcg];
#pragma unroll
    for (int j = 0; j < 8; ++j) Bs[bcg + j][brow] = bv[j];
    if (k0 + 32 < K) {  // prefetch next tiles into L2 (global_prefetch_b8)
      __builtin_prefetch(ag + 32, 0, 1);
      __builtin_prefetch(&Bm[(size_t)(k0 + 32 + brow) * N + n0 + bcg], 0, 1);
    }
    wait_async0();
    __syncthreads();

    // A fragment (16x32): lane m=l&15; K elems {hh*8..+7} U {16+hh*8..+7}
    const v16h a = make_frag(*(const v8h*)&As[wid * 16 + mfrag][hh * 8],
                             *(const v8h*)&As[wid * 16 + mfrag][16 + hh * 8]);
    // B fragments (32x16): lane n=l&15; K elems {hh*8..+7} U {16+hh*8..+7}
#pragma unroll
    for (int s = 0; s < 4; ++s) {
      const int ncol = s * 16 + mfrag;
      const v16h bf = make_frag(*(const v8h*)&Bs[ncol][hh * 8],
                                *(const v8h*)&Bs[ncol][16 + hh * 8]);
      acc[s] = __builtin_amdgcn_wmma_f32_16x16x32_f16(false, a, false, bf,
                                                      (short)0, acc[s], false, false);
    }
  }

  // C/D layout: VGPR r -> row M = r + 8*hh, col n = l&15
#pragma unroll
  for (int s = 0; s < 4; ++s) {
#pragma unroll
    for (int r = 0; r < 8; ++r) {
      const int row = m0 + wid * 16 + r + 8 * hh;
      const int col = n0 + s * 16 + mfrag;
      C[(size_t)row * N + col] = (OutT)acc[s][r];
    }
  }
}

// ---------------------------------------------------------------------------
// RoPE + reorder [b,s,h,d] -> [b,h,s,d]   (one thread per (b,s,h,pair))
// ---------------------------------------------------------------------------
__global__ __launch_bounds__(256) void rope_reorder(const _Float16* __restrict__ in,
                                                    const float* __restrict__ cosb,
                                                    const float* __restrict__ sinb,
                                                    _Float16* __restrict__ out,
                                                    int nheads) {
  const int idx = blockIdx.x * blockDim.x + threadIdx.x;
  const int total = BB * SS * nheads * HALF;
  if (idx >= total) return;
  const int i = idx & (HALF - 1);
  const int h = (idx >> 5) % nheads;
  const int s = (idx / (HALF * nheads)) % SS;
  const int b = idx / (HALF * nheads * SS);
  const float c  = cosb[s * HALF + i];
  const float sn = sinb[s * HALF + i];
  const _Float16* row = in + ((size_t)(b * SS + s)) * (nheads * HD) + h * HD;
  const float te = (float)row[2 * i];
  const float to = (float)row[2 * i + 1];
  _Float16* orow = out + (((size_t)(b * nheads + h)) * SS + s) * HD;
  orow[2 * i]     = (_Float16)(te * c - to * sn);
  orow[2 * i + 1] = (_Float16)(te * sn + to * c);
}

// V reorder [b,s,h,d] -> [b,h,s,d], f16 copy
__global__ __launch_bounds__(256) void v_reorder(const _Float16* __restrict__ in,
                                                 _Float16* __restrict__ out) {
  const int idx = blockIdx.x * blockDim.x + threadIdx.x;
  const int total = BB * SS * NKV * HD;
  if (idx >= total) return;
  const int d = idx & (HD - 1);
  const int h = (idx >> 6) % NKV;
  const int s = (idx / (HD * NKV)) % SS;
  const int b = idx / (HD * NKV * SS);
  out[(((size_t)(b * NKV + h)) * SS + s) * HD + d] =
      in[((size_t)(b * SS + s)) * (NKV * HD) + h * HD + d];
}

// ---------------------------------------------------------------------------
// Flash attention, one wave (32 threads) per (b, h, 16-row q tile).
// Q/K/V layout [b, head, s, 64] f16.  Online softmax, 32-key steps.
// Output written to attn_out [B*S, D] f16 (ready for the wo GEMM).
// ---------------------------------------------------------------------------
__global__ __launch_bounds__(32) void flash_attn(const _Float16* __restrict__ Q,
                                                 const _Float16* __restrict__ Kt,
                                                 const _Float16* __restrict__ V,
                                                 _Float16* __restrict__ Oo) {
  __shared__ __align__(16) _Float16 Ps[16][32];   // P tile (16 q x 32 k)
  __shared__ __align__(16) _Float16 Vst[64][32];  // V transposed: [d][k]

  const int lane  = threadIdx.x & 31;
  const int q0    = blockIdx.x * 16;
  const int h     = blockIdx.y;
  const int b     = blockIdx.z;
  const int hk    = h / (NH / NKV);
  const int mfrag = lane & 15;
  const int hh    = lane >> 4;
  const float scale = 0.125f;  // 1/sqrt(64)

  // Q fragments for the whole k loop (A-matrix layout)
  const size_t qbase = (((size_t)b * NH + h) * SS + q0) * HD;
  v16h qf[2];
#pragma unroll
  for (int c = 0; c < 2; ++c) {
    const _Float16* p = Q + qbase + (size_t)mfrag * HD + c * 32 + hh * 8;
    qf[c] = make_frag(*(const v8h*)p, *(const v8h*)(p + 16));
  }

  float rowmax[8], rowsum[8];
  v8f o[4] = {};
#pragma unroll
  for (int r = 0; r < 8; ++r) { rowmax[r] = -1e30f; rowsum[r] = 0.f; }

  const size_t kvbase = ((size_t)b * NKV + hk) * SS * HD;
  const int nsteps = (q0 + 15) / 32 + 1;  // causal bound

  for (int st = 0; st < nsteps; ++st) {
    const int k0 = st * 32;

    // S = Q K^T for two 16-key subtiles (B-frag: lane n=l&15 is key row,
    // K-dim elems d = c*32 + {(l>>4)*8..+7} U {16+(l>>4)*8..+7})
    v8f s1 = {}, s2 = {};
#pragma unroll
    for (int t = 0; t < 2; ++t) {
      const _Float16* krow = Kt + kvbase + (size_t)(k0 + t * 16 + mfrag) * HD;
#pragma unroll
      for (int c = 0; c < 2; ++c) {
        const v16h bf = make_frag(*(const v8h*)(krow + c * 32 + hh * 8),
                                  *(const v8h*)(krow + c * 32 + 16 + hh * 8));
        if (t == 0)
          s1 = __builtin_amdgcn_wmma_f32_16x16x32_f16(false, qf[c], false, bf,
                                                      (short)0, s1, false, false);
        else
          s2 = __builtin_amdgcn_wmma_f32_16x16x32_f16(false, qf[c], false, bf,
                                                      (short)0, s2, false, false);
      }
    }

    __syncthreads();  // previous iteration's Ps/Vst reads are done

    // Stage V rows k0..k0+31 transposed into LDS (lane = key row)
    {
      const _Float16* vrow = V + kvbase + (size_t)(k0 + lane) * HD;
#pragma unroll
      for (int j = 0; j < 8; ++j) {
        v8h vv = *(const v8h*)(vrow + j * 8);
#pragma unroll
        for (int e = 0; e < 8; ++e) Vst[j * 8 + e][lane] = vv[e];
      }
    }

    // Online softmax. C-layout: acc elem r -> row M=r+8*hh, col n=l&15.
    // Row reductions run over the 16 lanes of each half (xor masks < 16).
#pragma unroll
    for (int r = 0; r < 8; ++r) {
      const int qq = q0 + r + 8 * hh;
      float v1 = scale * s1[r]; if (k0 + mfrag      > qq) v1 = -1e30f;
      float v2 = scale * s2[r]; if (k0 + 16 + mfrag > qq) v2 = -1e30f;
      float t = fmaxf(v1, v2);
#pragma unroll
      for (int off = 1; off < 16; off <<= 1) t = fmaxf(t, __shfl_xor(t, off, 32));
      const float mnew  = fmaxf(rowmax[r], t);
      const float alpha = __expf(rowmax[r] - mnew);
      const float e1 = __expf(v1 - mnew);
      const float e2 = __expf(v2 - mnew);
      float sl = e1 + e2;
#pragma unroll
      for (int off = 1; off < 16; off <<= 1) sl += __shfl_xor(sl, off, 32);
      rowsum[r] = rowsum[r] * alpha + sl;
      rowmax[r] = mnew;
      Ps[r + 8 * hh][mfrag]      = (_Float16)e1;
      Ps[r + 8 * hh][16 + mfrag] = (_Float16)e2;
#pragma unroll
      for (int nt = 0; nt < 4; ++nt) o[nt][r] *= alpha;
    }

    __syncthreads();  // Ps and Vst visible

    // O += P (16x32, A-layout from LDS) * V (32x64, B-frags from Vst)
    const v16h pf = make_frag(*(const v8h*)&Ps[mfrag][hh * 8],
                              *(const v8h*)&Ps[mfrag][16 + hh * 8]);
#pragma unroll
    for (int nt = 0; nt < 4; ++nt) {
      const v16h vf = make_frag(*(const v8h*)&Vst[nt * 16 + mfrag][hh * 8],
                                *(const v8h*)&Vst[nt * 16 + mfrag][16 + hh * 8]);
      o[nt] = __builtin_amdgcn_wmma_f32_16x16x32_f16(false, pf, false, vf,
                                                     (short)0, o[nt], false, false);
    }
  }

  // Normalize and write to attn_out [B*S, D] (col = h*64 + nt*16 + n)
#pragma unroll
  for (int r = 0; r < 8; ++r) {
    const int M = r + 8 * hh;
    const float inv = 1.0f / rowsum[r];
    _Float16* orow = Oo + ((size_t)(b * SS + q0 + M)) * DD + h * HD + mfrag;
#pragma unroll
    for (int nt = 0; nt < 4; ++nt) orow[nt * 16] = (_Float16)(o[nt][r] * inv);
  }
}

// ---------------------------------------------------------------------------
// Host orchestration
// ---------------------------------------------------------------------------
static inline size_t align256(size_t x) { return (x + 255) & ~(size_t)255; }

extern "C" void kernel_launch(void* const* d_in, const int* in_sizes, int n_in,
                              void* d_out, int out_size, void* d_ws, size_t ws_size,
                              hipStream_t stream) {
  (void)in_sizes; (void)n_in; (void)out_size; (void)ws_size;

  const float* x    = (const float*)d_in[0];
  const float* cosb = (const float*)d_in[1];
  const float* sinb = (const float*)d_in[2];
  const float* wq   = (const float*)d_in[3];
  const float* wk   = (const float*)d_in[4];
  const float* wv   = (const float*)d_in[5];
  const float* wo   = (const float*)d_in[6];
  float* out = (float*)d_out;

  const size_t M  = (size_t)BB * SS;       // 4096 token rows
  const size_t nX  = M * DD;               // x / q elements
  const size_t nW  = (size_t)DD * DD;      // wq / wo elements
  const size_t nKV = M * (size_t)NKV * HD; // k / v elements
  const size_t nWk = (size_t)DD * NKV * HD;

  // carve workspace (~100 MB)
  char* ws = (char*)d_ws;
  size_t off = 0;
  auto carve = [&](size_t bytes) { char* p = ws + off; off += align256(bytes); return p; };
  _Float16* xh    = (_Float16*)carve(nX  * 2);
  _Float16* wqh   = (_Float16*)carve(nW  * 2);
  _Float16* wkh   = (_Float16*)carve(nWk * 2);
  _Float16* wvh   = (_Float16*)carve(nWk * 2);
  _Float16* woh   = (_Float16*)carve(nW  * 2);
  _Float16* qproj = (_Float16*)carve(nX  * 2);
  _Float16* kproj = (_Float16*)carve(nKV * 2);
  _Float16* vproj = (_Float16*)carve(nKV * 2);
  _Float16* qr    = (_Float16*)carve(nX  * 2);
  _Float16* kr    = (_Float16*)carve(nKV * 2);
  _Float16* vr    = (_Float16*)carve(nKV * 2);
  _Float16* ao    = (_Float16*)carve(nX  * 2);

  auto cast = [&](const float* src, _Float16* dst, size_t n) {
    int n4 = (int)(n / 4);
    cast_f32_to_f16<<<(n4 + 255) / 256, 256, 0, stream>>>(src, dst, n4);
  };
  cast(x,  xh,  nX);
  cast(wq, wqh, nW);
  cast(wk, wkh, nWk);
  cast(wv, wvh, nWk);
  cast(wo, woh, nW);

  // QKV projections (f16 out)
  wmma_gemm<_Float16><<<dim3(DD / 64, (int)M / 128), 256, 0, stream>>>(xh, wqh, qproj,
                                                                      (int)M, DD, DD);
  wmma_gemm<_Float16><<<dim3((NKV * HD) / 64, (int)M / 128), 256, 0, stream>>>(xh, wkh, kproj,
                                                                               (int)M, NKV * HD, DD);
  wmma_gemm<_Float16><<<dim3((NKV * HD) / 64, (int)M / 128), 256, 0, stream>>>(xh, wvh, vproj,
                                                                               (int)M, NKV * HD, DD);

  // RoPE + head-major reorder
  {
    int tq = BB * SS * NH * HALF;
    rope_reorder<<<(tq + 255) / 256, 256, 0, stream>>>(qproj, cosb, sinb, qr, NH);
    int tk = BB * SS * NKV * HALF;
    rope_reorder<<<(tk + 255) / 256, 256, 0, stream>>>(kproj, cosb, sinb, kr, NKV);
    int tv = BB * SS * NKV * HD;
    v_reorder<<<(tv + 255) / 256, 256, 0, stream>>>(vproj, vr);
  }

  // Flash attention: one wave per (q-tile, head, batch)
  flash_attn<<<dim3(SS / 16, NH, BB), 32, 0, stream>>>(qr, kr, vr, ao);

  // Output projection (f32 out)
  wmma_gemm<float><<<dim3(DD / 64, (int)M / 128), 256, 0, stream>>>(ao, woh, out,
                                                                    (int)M, DD, DD);
}
